// E62bDecaySelectiveCell_4887672783138
// MI455X (gfx1250) — matrix operand order repaired and checked by reading, not compiled, not run on hardware
//
#include <hip/hip_runtime.h>
#include <hip/hip_bf16.h>

typedef __attribute__((ext_vector_type(16))) __bf16 v16bf;
typedef __attribute__((ext_vector_type(8)))  __bf16 v8bf;
typedef __attribute__((ext_vector_type(4)))  __bf16 v4bf;
typedef __attribute__((ext_vector_type(8)))  float  v8f;

#define TT 2048
#define BBATCH 16
#define DD 1024
#define MM (TT*BBATCH)
#define LDS_STRIDE 40                 // 32 halfs + 8 pad (80B row stride, 16B aligned)
#define A_ELEMS (128 * LDS_STRIDE)
#define B_ELEMS (3 * 64 * LDS_STRIDE)
#define KSTEPS (DD / 32)

// Native hardware f32->bf16 convert (v_cvt_pk_bf16_f32 on gfx1250)
__device__ __forceinline__ __bf16 cvt_bf16(float f) {
  return static_cast<__bf16>(f);
}

__device__ __forceinline__ float sigmoid_f(float x) {
  return 1.0f / (1.0f + __expf(-x));
}

__device__ __forceinline__ float tanh_f(float x) {
  x = fminf(fmaxf(x, -15.0f), 15.0f);
  float e = __expf(2.0f * x);
  return (e - 1.0f) / (e + 1.0f);
}

// Two 16B LDS loads -> one 16-half fragment
__device__ __forceinline__ v16bf ld_frag(const __bf16* p0, const __bf16* p1) {
  v8bf lo = *(const v8bf*)p0;
  v8bf hi = *(const v8bf*)p1;
  return __builtin_shufflevector(lo, hi, 0,1,2,3,4,5,6,7,8,9,10,11,12,13,14,15);
}

// Phase 1: fused 3-gate GEMM (x @ W^T) with bf16 WMMA, f32 accum.
// Block tile: 128 (M) x 64 (N); 512 threads = 16 waves in 4(M)x4(N) grid;
// each wave: 32(M) x 16(N) per gate = 6 accumulators.
// Pipeline: double-buffered LDS; next tile's 5 global b128 loads issue before
// the WMMA block, cvt+ds_store after it; one barrier per K-step.
__global__ __launch_bounds__(512)
void gates_gemm_kernel(const float* __restrict__ x,
                       const float* __restrict__ Wa,
                       const float* __restrict__ Wk,
                       const float* __restrict__ Wv,
                       const float* __restrict__ ba,
                       const float* __restrict__ bk,
                       const float* __restrict__ bv,
                       float* __restrict__ retain_out,
                       float* __restrict__ write_out) {
  __shared__ __bf16 Abuf[2 * A_ELEMS];   // 2 x (128 rows x 32 K)
  __shared__ __bf16 Bbuf[2 * B_ELEMS];   // 2 x (3 gates x 64 N x 32 K)

  const int tid      = threadIdx.x;
  const int lane     = tid & 31;
  const int wid      = tid >> 5;       // 0..15
  const int wm       = wid & 3;        // 0..3 -> M offset wm*32
  const int wn       = wid >> 2;       // 0..3 -> N offset wn*16
  const int lane15   = lane & 15;
  const int laneHalf = lane >> 4;

  const int m0 = blockIdx.x * 128;
  const int n0 = blockIdx.y * 64;

  const float* Ws[3] = {Wa, Wk, Wv};

  // Per-thread staging coordinates (constant over K-loop)
  const int arow[2] = { (tid + 0*512) >> 3, (tid + 1*512) >> 3 };  // 0..127
  const int brow    = tid >> 3;                                    // 0..63
  const int ccol    = (tid & 7) * 4;                               // 0..28

  v8f acc[2][3];                       // [tm][gate]
  #pragma unroll
  for (int i = 0; i < 2; ++i)
    #pragma unroll
    for (int g = 0; g < 3; ++g)
      #pragma unroll
      for (int r = 0; r < 8; ++r) acc[i][g][r] = 0.0f;

  const int klo_a = laneHalf * 8;    // A frag K base: 0 or 8 (plus +16 for hi half)
  const int klo_b = laneHalf * 16;   // unused (0 or 16); B frag also uses klo_a

  float4 areg[2];
  float4 breg[3];

  auto issue_loads = [&](int kk) {
    #pragma unroll
    for (int j = 0; j < 2; ++j)
      areg[j] = *(const float4*)(x + (size_t)(m0 + arow[j]) * DD + kk + ccol);
    #pragma unroll
    for (int g = 0; g < 3; ++g)
      breg[g] = *(const float4*)(Ws[g] + (size_t)(n0 + brow) * DD + kk + ccol);
  };

  auto store_lds = [&](int buf) {
    __bf16* Ab = Abuf + buf * A_ELEMS;
    __bf16* Bb = Bbuf + buf * B_ELEMS;
    #pragma unroll
    for (int j = 0; j < 2; ++j) {
      v4bf hv;
      hv[0] = cvt_bf16(areg[j].x); hv[1] = cvt_bf16(areg[j].y);
      hv[2] = cvt_bf16(areg[j].z); hv[3] = cvt_bf16(areg[j].w);
      *(v4bf*)(&Ab[arow[j] * LDS_STRIDE + ccol]) = hv;
    }
    #pragma unroll
    for (int g = 0; g < 3; ++g) {
      v4bf hv;
      hv[0] = cvt_bf16(breg[g].x); hv[1] = cvt_bf16(breg[g].y);
      hv[2] = cvt_bf16(breg[g].z); hv[3] = cvt_bf16(breg[g].w);
      *(v4bf*)(&Bb[(g * 64 + brow) * LDS_STRIDE + ccol]) = hv;
    }
  };

  auto compute = [&](int buf) {
    const __bf16* Ab = Abuf + buf * A_ELEMS;
    const __bf16* Bb = Bbuf + buf * B_ELEMS;
    // Load ALL fragments first -> single dscnt wait, then 6 back-to-back WMMAs
    v16bf afrag[2];
    v16bf bfrag[3];
    #pragma unroll
    for (int tm = 0; tm < 2; ++tm) {
      const __bf16* rp = &Ab[(wm * 32 + tm * 16 + lane15) * LDS_STRIDE];
      afrag[tm] = ld_frag(rp + klo_a, rp + 16 + klo_a);
    }
    const int nrow = wn * 16 + lane15;
    #pragma unroll
    for (int g = 0; g < 3; ++g) {
      const __bf16* bp = &Bb[(g * 64 + nrow) * LDS_STRIDE];
      bfrag[g] = ld_frag(bp + klo_a, bp + 16 + klo_a);
    }
    #pragma unroll
    for (int g = 0; g < 3; ++g)
      #pragma unroll
      for (int tm = 0; tm < 2; ++tm)
        acc[tm][g] = __builtin_amdgcn_wmma_f32_16x16x32_bf16(
            false, afrag[tm], false, bfrag[g], (short)0, acc[tm][g],
            false, false);
  };

  // ---- pipeline ----
  issue_loads(0);
  store_lds(0);
  __syncthreads();

  for (int i = 0; i < KSTEPS; ++i) {
    const bool has_next = (i + 1) < KSTEPS;
    if (has_next) issue_loads((i + 1) * 32);   // 5 global b128 loads in flight
    compute(i & 1);
    if (has_next) store_lds((i + 1) & 1);      // 10 cvt_pk + 5 ds_store_b64
    __syncthreads();
  }

  // --- epilogue: bias + sigmoid/sigmoid/tanh, combine, store retain & write ---
  // C/D layout: VGPR r, lane l -> element (m = r + (l>=16)*8, n = l&15)
  {
    const int col = n0 + wn * 16 + lane15;
    const float biasA = ba[col];
    const float biasK = bk[col];
    const float biasV = bv[col];
    #pragma unroll
    for (int tm = 0; tm < 2; ++tm) {
      const int rbase = m0 + wm * 32 + tm * 16 + laneHalf * 8;
      #pragma unroll
      for (int r = 0; r < 8; ++r) {
        const size_t o = (size_t)(rbase + r) * DD + col;
        float aG = sigmoid_f(acc[tm][0][r] + biasA);
        float kG = sigmoid_f(acc[tm][1][r] + biasK);
        float vG = tanh_f   (acc[tm][2][r] + biasV);
        retain_out[o] = aG * (1.0f - kG);   // staged in output[t] slot
        write_out[o]  = kG * vG;            // staged in h[t+1] slot
      }
    }
  }
}

// Phase 2: diagonal linear scan + SiLU-squared output, in place over d_out.
// One lane per (b,d) channel; loads coalesced across lanes at each t.
__global__ __launch_bounds__(256)
void scan_silu_kernel(const float* __restrict__ h0, float* __restrict__ out) {
  const int BD = BBATCH * DD;
  const int c = blockIdx.x * blockDim.x + threadIdx.x;   // 0 .. BD-1
  float* retain = out;                          // [T, BD] -> becomes output
  float* hbuf   = out + (size_t)TT * BD;        // [T+1, BD]; write[t] staged at h[t+1]

  float h = h0[c];
  hbuf[c] = h;                                  // h[0] = h0
  #pragma unroll 4
  for (int t = 0; t < TT; ++t) {
    const size_t o = (size_t)t * BD + c;
    float a = retain[o];
    float b = hbuf[o + BD];
    h = fmaf(a, h, b);                          // h_t = a*h_{t-1} + b
    float s = 1.0f / (1.0f + __expf(-h));
    retain[o]   = h * h * s;                    // output_t = h^2 * sigmoid(h)
    hbuf[o + BD] = h;                           // h_t
  }
}

extern "C" void kernel_launch(void* const* d_in, const int* in_sizes, int n_in,
                              void* d_out, int out_size, void* d_ws, size_t ws_size,
                              hipStream_t stream) {
  const float* x  = (const float*)d_in[0];
  const float* h0 = (const float*)d_in[1];
  const float* Wa = (const float*)d_in[2];
  const float* ba = (const float*)d_in[3];
  const float* Wk = (const float*)d_in[4];
  const float* bk = (const float*)d_in[5];
  const float* Wv = (const float*)d_in[6];
  const float* bv = (const float*)d_in[7];

  float* out    = (float*)d_out;
  float* retain = out;                                                  // output region
  float* writeb = out + (size_t)TT * BBATCH * DD + (size_t)BBATCH * DD; // h[1..T] slots

  dim3 g1(MM / 128, DD / 64, 1);
  gates_gemm_kernel<<<g1, 512, 0, stream>>>(x, Wa, Wk, Wv, ba, bk, bv,
                                            retain, writeb);

  const int BD = BBATCH * DD;
  scan_silu_kernel<<<BD / 256, 256, 0, stream>>>(h0, out);
}
